// HRFPN_MSFT_50714973831298
// MI455X (gfx1250) — matrix operand order repaired and checked by reading, not compiled, not run on hardware
//
// CDNA5 (gfx1250 / MI455X) dual-stream cross-attention transformer, bf16 WMMA.
//
// Roofline: ~1.24 TFLOP of GEMM/attention math per call; unique data (~120MB
// activations+weights) fits in the 192MB L2, HBM @23.3TB/s is irrelevant ->
// compute bound. All matmuls routed through v_wmma_f32_16x16x32_bf16 (wave32),
// f32 accumulate.
//
// v2 changes vs v1:
//  - Weights are converted fp32->bf16 AND transposed to [N][K] ONCE per call
//    (amortized over all 8200 rows), so the GEMM hot loop is pure b128
//    global->LDS staging + WMMA (no per-tile cvt / ds_store_b16 scatter).
//  - GEMM block tile 128x128, wave tile 32x64 -> 8 WMMAs per K-step per wave.
//  - Double-buffered LDS (40KB of the 320KB WGP pool), 1 barrier per K-step.
#include <hip/hip_runtime.h>
#include <cmath>

typedef __bf16 bf16;
typedef __attribute__((ext_vector_type(16))) __bf16 v16bf;
typedef __attribute__((ext_vector_type(8)))  __bf16 v8bf;
typedef __attribute__((ext_vector_type(4)))  __bf16 v4bf;
typedef __attribute__((ext_vector_type(8)))  float  v8f;

#define XDEPTH   6
#define XDIM     512
#define XHEADS   8
#define XDHEAD   64
#define XMLP     2048
#define XBATCH   8
#define XSEQ     1025
#define XTOK     (XBATCH * XSEQ)       // 8200 tokens per stream
#define ATT_SCALE 0.04419417382415922f // 512^-0.5 (reference scales by dim^-0.5)

// ---------------- WMMA helper ----------------
__device__ __forceinline__ v8f wmma_bf16(v16bf a, v16bf b, v8f c) {
  return __builtin_amdgcn_wmma_f32_16x16x32_bf16(false, a, false, b, (short)0, c,
                                                 false, false);
}

// ---------------- Weight convert+transpose: W[K][N] f32 -> Wt[N][K] bf16 ----
__global__ __launch_bounds__(256) void convert_transpose_kernel(
    const float* __restrict__ W, bf16* __restrict__ Wt, int K, int N) {
  __shared__ bf16 tile[32][36];
  const int k0 = blockIdx.x * 32;
  const int n0 = blockIdx.y * 32;
  const int r  = threadIdx.x >> 3;        // 0..31
  const int c4 = (threadIdx.x & 7) * 4;   // 0..28
  float4 f = *(const float4*)(W + (size_t)(k0 + r) * N + n0 + c4);
  tile[r][c4 + 0] = (bf16)f.x;
  tile[r][c4 + 1] = (bf16)f.y;
  tile[r][c4 + 2] = (bf16)f.z;
  tile[r][c4 + 3] = (bf16)f.w;
  __syncthreads();
  v4bf o;
#pragma unroll
  for (int j = 0; j < 4; ++j) o[j] = tile[c4 + j][r];
  *(v4bf*)(Wt + (size_t)(n0 + r) * K + k0 + c4) = o;
}

// ---------------- LayerNorm (f32 -> bf16), one wave per token --------------
__global__ __launch_bounds__(256) void ln_to_bf16_kernel(
    const float* __restrict__ x, const float* __restrict__ g,
    const float* __restrict__ b, bf16* __restrict__ out, int T) {
  const int lane = threadIdx.x & 31;
  const int wave = threadIdx.x >> 5;
  const int t = blockIdx.x * 8 + wave;
  if (t >= T) return;

  const float4* rp = (const float4*)(x + (size_t)t * XDIM);
  float4 r[4];
#pragma unroll
  for (int i = 0; i < 4; ++i) r[i] = rp[lane * 4 + i];

  float s1 = 0.f, s2 = 0.f;
#pragma unroll
  for (int i = 0; i < 4; ++i) {
    s1 += r[i].x + r[i].y + r[i].z + r[i].w;
    s2 += r[i].x * r[i].x + r[i].y * r[i].y + r[i].z * r[i].z + r[i].w * r[i].w;
  }
#pragma unroll
  for (int m = 16; m >= 1; m >>= 1) {
    s1 += __shfl_xor(s1, m, 32);
    s2 += __shfl_xor(s2, m, 32);
  }
  const float mean = s1 * (1.0f / XDIM);
  const float var  = s2 * (1.0f / XDIM) - mean * mean;
  const float rstd = rsqrtf(var + 1e-5f);

  __align__(16) bf16 ob[16];
#pragma unroll
  for (int i = 0; i < 4; ++i) {
    float4 gg = ((const float4*)g)[lane * 4 + i];
    float4 bb = ((const float4*)b)[lane * 4 + i];
    ob[i * 4 + 0] = (bf16)((r[i].x - mean) * rstd * gg.x + bb.x);
    ob[i * 4 + 1] = (bf16)((r[i].y - mean) * rstd * gg.y + bb.y);
    ob[i * 4 + 2] = (bf16)((r[i].z - mean) * rstd * gg.z + bb.z);
    ob[i * 4 + 3] = (bf16)((r[i].w - mean) * rstd * gg.w + bb.w);
  }
  v8bf* op = (v8bf*)(out + (size_t)t * XDIM + lane * 16);
  op[0] = *(const v8bf*)&ob[0];
  op[1] = *(const v8bf*)&ob[8];
}

// ---------------- GEMM: out = A(bf16, MxK) @ Wt(bf16, [N][K]) --------------
// Block tile 128x128, 8 waves (4x2), wave tile 32x64 (2x4 WMMA), K-step 32,
// double-buffered LDS. Epilogue fuses bias / GELU / residual.
#define GBM 128
#define GBN 128
#define GBK 32
#define LSTR 40  // LDS row stride (elems): 80B, 16B aligned, conflict pad

__global__ __launch_bounds__(256) void gemm_bf16_wmma_kernel(
    const bf16* __restrict__ A, int lda, const bf16* __restrict__ Wt, int ldwt,
    const float* __restrict__ bias, const float* __restrict__ resid,
    float* __restrict__ outF, bf16* __restrict__ outB, int ldo, int M, int Kdim,
    int gelu_flag) {
  __shared__ __align__(16) bf16 Asmem[2][GBM][LSTR];
  __shared__ __align__(16) bf16 Bsmem[2][GBN][LSTR];

  const int tid  = threadIdx.x;
  const int lane = tid & 31;
  const int wid  = tid >> 5;
  const int wm   = wid >> 1;  // 0..3 : 32-row strip
  const int wn   = wid & 1;   // 0..1 : 64-col strip
  const int m0   = blockIdx.x * GBM;
  const int n0   = blockIdx.y * GBN;

  v8f acc[2][4];
#pragma unroll
  for (int i = 0; i < 2; ++i)
#pragma unroll
    for (int j = 0; j < 4; ++j)
#pragma unroll
      for (int r = 0; r < 8; ++r) acc[i][j][r] = 0.f;

  // tile staging: 512 b128 chunks per 128x32 tile; thread t owns chunks t, t+256
  v8bf av[2], bv[2];
  auto load_tiles = [&](int k0) {
#pragma unroll
    for (int i = 0; i < 2; ++i) {
      const int c   = tid + i * 256;
      const int row = c >> 2;
      const int ch  = (c & 3) * 8;
      const int gm  = m0 + row;
      if (gm < M) {
        av[i] = *(const v8bf*)(A + (size_t)gm * lda + k0 + ch);
      } else {
#pragma unroll
        for (int j = 0; j < 8; ++j) av[i][j] = (bf16)0.0f;
      }
      const bf16* wp = Wt + (size_t)(n0 + row) * ldwt + k0 + ch;
      bv[i] = *(const v8bf*)wp;
      if (i == 0) __builtin_prefetch(wp + GBK, 0, 1);  // global_prefetch_b8
    }
  };
  auto store_tiles = [&](int buf) {
#pragma unroll
    for (int i = 0; i < 2; ++i) {
      const int c   = tid + i * 256;
      const int row = c >> 2;
      const int ch  = (c & 3) * 8;
      *(v8bf*)&Asmem[buf][row][ch] = av[i];
      *(v8bf*)&Bsmem[buf][row][ch] = bv[i];
    }
  };

  const int nk = Kdim / GBK;
  load_tiles(0);
  int buf = 0;
  for (int kt = 0; kt < nk; ++kt) {
    store_tiles(buf);
    __syncthreads();
    if (kt + 1 < nk) load_tiles((kt + 1) * GBK);  // overlap with WMMAs below

    v16bf afr[2], bfr[4];
    const int kb = (lane >> 4) * 8;  // A lane layout: K chunks kb, 16+kb
#pragma unroll
    for (int tm = 0; tm < 2; ++tm) {
      const int ar = wm * 32 + tm * 16 + (lane & 15);
      *(v8bf*)&afr[tm]         = *(const v8bf*)&Asmem[buf][ar][kb];
      *(((v8bf*)&afr[tm]) + 1) = *(const v8bf*)&Asmem[buf][ar][16 + kb];
    }
    const int kb2 = (lane >> 4) * 8;  // B lane layout: K chunks kb2, 16+kb2
#pragma unroll
    for (int tn = 0; tn < 4; ++tn) {
      const int bc = wn * 64 + tn * 16 + (lane & 15);
      *(v8bf*)&bfr[tn]         = *(const v8bf*)&Bsmem[buf][bc][kb2];
      *(((v8bf*)&bfr[tn]) + 1) = *(const v8bf*)&Bsmem[buf][bc][kb2 + 16];
    }
#pragma unroll
    for (int tm = 0; tm < 2; ++tm)
#pragma unroll
      for (int tn = 0; tn < 4; ++tn)
        acc[tm][tn] = wmma_bf16(afr[tm], bfr[tn], acc[tm][tn]);
    buf ^= 1;
  }

  // epilogue: C layout lane = col, vgpr r = row
#pragma unroll
  for (int tm = 0; tm < 2; ++tm)
#pragma unroll
    for (int tn = 0; tn < 4; ++tn)
#pragma unroll
      for (int r = 0; r < 8; ++r) {
        const int m = m0 + wm * 32 + tm * 16 + (lane >> 4) * 8 + r;
        const int n = n0 + wn * 64 + tn * 16 + (lane & 15);
        if (m >= M) continue;
        float v = acc[tm][tn][r];
        if (bias) v += bias[n];
        if (outF) {  // f32 output with residual add (N==512 path)
          v += resid[(size_t)m * XDIM + n];
          outF[(size_t)m * ldo + n] = v;
        } else {
          if (gelu_flag) v = 0.5f * v * (1.0f + erff(v * 0.70710678f));
          outB[(size_t)m * ldo + n] = (bf16)v;
        }
      }
}

// ---------------- Flash attention (bf16 WMMA, f32 softmax) ----------------
#define AT_MT 128
#define AT_KT 32

__global__ __launch_bounds__(256) void attention_wmma_kernel(
    const bf16* __restrict__ Q, int ldq, const bf16* __restrict__ Km, int ldk,
    const bf16* __restrict__ V, int ldv, bf16* __restrict__ O, int ldo,
    int seq) {
  __shared__ __align__(16) bf16 Klds[AT_KT][XDHEAD + 8];   // [key][d]
  __shared__ __align__(16) bf16 Vlds[XDHEAD][AT_KT + 8];   // [d][key]
  __shared__ __align__(16) bf16 Plds[8][16][AT_KT + 8];    // per-wave P scratch

  const int tid = threadIdx.x, lane = tid & 31, w = tid >> 5;
  const int bh = blockIdx.y;
  const int bb = bh >> 3;
  const int h  = bh & (XHEADS - 1);
  const int hoff = h * XDHEAD;
  const int rowbase = blockIdx.x * AT_MT + w * 16;

  v16bf qf[2];
  {
    const int r  = rowbase + (lane & 15);
    const int kb = (lane >> 4) * 8;
    const bool ok = r < seq;
    const bf16* qp = Q + ((size_t)(bb * seq) + (ok ? r : 0)) * ldq + hoff;
#pragma unroll
    for (int f = 0; f < 2; ++f) {
      v8bf c0, c1;
      if (ok) {
        c0 = *(const v8bf*)(qp + f * 32 + kb);
        c1 = *(const v8bf*)(qp + f * 32 + 16 + kb);
      } else {
#pragma unroll
        for (int j = 0; j < 8; ++j) { c0[j] = (bf16)0.f; c1[j] = (bf16)0.f; }
      }
      *(v8bf*)&qf[f]         = c0;
      *(((v8bf*)&qf[f]) + 1) = c1;
    }
  }

  float mrow[8], lsum[8];
#pragma unroll
  for (int r = 0; r < 8; ++r) { mrow[r] = -1e30f; lsum[r] = 0.f; }
  v8f o[4];
#pragma unroll
  for (int d = 0; d < 4; ++d)
#pragma unroll
    for (int r = 0; r < 8; ++r) o[d][r] = 0.f;

  for (int j0 = 0; j0 < seq; j0 += AT_KT) {
    {
      const int key = tid >> 3;
      const int ch  = (tid & 7) * 8;
      const int gj  = j0 + key;
      v8bf kv, vv;
      if (gj < seq) {
        kv = *(const v8bf*)(Km + ((size_t)(bb * seq) + gj) * ldk + hoff + ch);
        vv = *(const v8bf*)(V + ((size_t)(bb * seq) + gj) * ldv + hoff + ch);
      } else {
#pragma unroll
        for (int j = 0; j < 8; ++j) { kv[j] = (bf16)0.f; vv[j] = (bf16)0.f; }
      }
      *(v8bf*)&Klds[key][ch] = kv;
#pragma unroll
      for (int j = 0; j < 8; ++j) Vlds[ch + j][key] = vv[j];
    }
    __syncthreads();

    v8f s[2];
#pragma unroll
    for (int jt = 0; jt < 2; ++jt) {
#pragma unroll
      for (int r = 0; r < 8; ++r) s[jt][r] = 0.f;
#pragma unroll
      for (int ks = 0; ks < 2; ++ks) {
        v16bf kf;
        const int kc = jt * 16 + (lane & 15);
        const int dc = ks * 32 + (lane >> 4) * 8;
        *(v8bf*)&kf         = *(const v8bf*)&Klds[kc][dc];
        *(((v8bf*)&kf) + 1) = *(const v8bf*)&Klds[kc][dc + 16];
        s[jt] = wmma_bf16(qf[ks], kf, s[jt]);
      }
    }

    float corr[8];
    const int c0i = j0 + (lane & 15);
    const int c1i = c0i + 16;
#pragma unroll
    for (int r = 0; r < 8; ++r) {
      float s0 = s[0][r] * ATT_SCALE;
      float s1 = s[1][r] * ATT_SCALE;
      if (c0i >= seq) s0 = -1e30f;
      if (c1i >= seq) s1 = -1e30f;
      float mx = fmaxf(s0, s1);
#pragma unroll
      for (int msk = 8; msk >= 1; msk >>= 1) mx = fmaxf(mx, __shfl_xor(mx, msk, 32));
      const float nm = fmaxf(mrow[r], mx);
      const float cc = __expf(mrow[r] - nm);
      const float p0 = __expf(s0 - nm);
      const float p1 = __expf(s1 - nm);
      float ps = p0 + p1;
#pragma unroll
      for (int msk = 8; msk >= 1; msk >>= 1) ps += __shfl_xor(ps, msk, 32);
      lsum[r] = lsum[r] * cc + ps;
      mrow[r] = nm;
      corr[r] = cc;
      const int pr = (lane >> 4) * 8 + r;
      Plds[w][pr][lane & 15]        = (bf16)p0;
      Plds[w][pr][16 + (lane & 15)] = (bf16)p1;
    }
#pragma unroll
    for (int dt = 0; dt < 4; ++dt)
#pragma unroll
      for (int r = 0; r < 8; ++r) o[dt][r] *= corr[r];

    // wave-local LDS RAW fence (CDNA5 split counter)
    asm volatile("s_wait_dscnt 0x0" ::: "memory");

    v16bf pf;
    {
      const int pm = lane & 15;
      const int kb = (lane >> 4) * 8;
      *(v8bf*)&pf         = *(const v8bf*)&Plds[w][pm][kb];
      *(((v8bf*)&pf) + 1) = *(const v8bf*)&Plds[w][pm][16 + kb];
    }
#pragma unroll
    for (int dt = 0; dt < 4; ++dt) {
      v16bf vf;
      const int dc = dt * 16 + (lane & 15);
      const int kc = (lane >> 4) * 8;
      *(v8bf*)&vf         = *(const v8bf*)&Vlds[dc][kc];
      *(((v8bf*)&vf) + 1) = *(const v8bf*)&Vlds[dc][kc + 16];
      o[dt] = wmma_bf16(pf, vf, o[dt]);
    }
    __syncthreads();
  }

#pragma unroll
  for (int dt = 0; dt < 4; ++dt)
#pragma unroll
    for (int r = 0; r < 8; ++r) {
      const int row = rowbase + (lane >> 4) * 8 + r;
      if (row >= seq) continue;
      const int d = dt * 16 + (lane & 15);
      O[((size_t)(bb * seq) + row) * ldo + hoff + d] = (bf16)(o[dt][r] / lsum[r]);
    }
}

// ---------------- host orchestration ----------------
static void launch_ln(const float* x, const float* g, const float* b, bf16* out,
                      hipStream_t s) {
  ln_to_bf16_kernel<<<(XTOK + 7) / 8, 256, 0, s>>>(x, g, b, out, XTOK);
}
static void launch_gemm(const bf16* A, int lda, const bf16* Wt, int ldwt,
                        const float* bias, const float* resid, float* outF,
                        bf16* outB, int ldo, int N, int K, int gelu,
                        hipStream_t s) {
  dim3 grid((XTOK + GBM - 1) / GBM, N / GBN);
  gemm_bf16_wmma_kernel<<<grid, 256, 0, s>>>(A, lda, Wt, ldwt, bias, resid,
                                             outF, outB, ldo, XTOK, K, gelu);
}
static void launch_attn(const bf16* Q, int ldq, const bf16* K, int ldk,
                        const bf16* V, int ldv, bf16* O, hipStream_t s) {
  dim3 grid((XSEQ + AT_MT - 1) / AT_MT, XBATCH * XHEADS);
  attention_wmma_kernel<<<grid, 256, 0, s>>>(Q, ldq, K, ldk, V, ldv, O, XDIM,
                                             XSEQ);
}

extern "C" void kernel_launch(void* const* d_in, const int* in_sizes, int n_in,
                              void* d_out, int out_size, void* d_ws,
                              size_t ws_size, hipStream_t stream) {
  (void)in_sizes; (void)n_in; (void)out_size; (void)ws_size;
  const size_t TD = (size_t)XTOK * XDIM;

  char* wsp = (char*)d_ws;
  auto carve = [&](size_t bytes) {
    void* p = wsp;
    wsp += (bytes + 255) & ~(size_t)255;
    return p;
  };
  float* xA  = (float*)carve(TD * 4);
  float* xB  = (float*)carve(TD * 4);
  float* xA2 = (float*)carve(TD * 4);
  bf16* lnq  = (bf16*)carve(TD * 2);
  bf16* lnkv = (bf16*)carve(TD * 2);
  bf16* qkv  = (bf16*)carve((size_t)XTOK * 3 * XDIM * 2);
  bf16* kvb  = (bf16*)carve((size_t)XTOK * 2 * XDIM * 2);
  bf16* attn = (bf16*)carve(TD * 2);
  bf16* hid  = (bf16*)carve((size_t)XTOK * XMLP * 2);
  const size_t WLAYER = 4194304;  // bf16 elems of all 7 matrices per layer
  bf16* warena = (bf16*)carve((size_t)2 * XDEPTH * WLAYER * 2);

  hipMemcpyAsync(xA, d_in[0], TD * 4, hipMemcpyDeviceToDevice, stream);
  hipMemcpyAsync(xB, d_in[1], TD * 4, hipMemcpyDeviceToDevice, stream);

  enum { LN1G, LN1B, WQKV, WOW, WOB, LN2G, LN2B, WQm, WKV, W2OW, W2OB,
         LN3G, LN3B, WF1, BF1m, WF2, BF2m };
  const float* prm[2][17];
  for (int s = 0; s < 2; ++s)
    for (int k = 0; k < 17; ++k) prm[s][k] = (const float*)d_in[2 + s * 17 + k];

  // weight arena layout per (stream, layer): [qkv, ow, q, kv, w2ow, f1, f2]
  struct WDesc { int pidx, K, N; size_t off; };
  const WDesc WD[7] = {
      {WQKV, XDIM, 3 * XDIM, 0},       {WOW, XDIM, XDIM, 786432},
      {WQm, XDIM, XDIM, 1048576},      {WKV, XDIM, 2 * XDIM, 1310720},
      {W2OW, XDIM, XDIM, 1835008},     {WF1, XDIM, XMLP, 2097152},
      {WF2, XMLP, XDIM, 3145728}};
  auto wt = [&](int s, int i, int j) {
    return warena + (size_t)(s * XDEPTH + i) * WLAYER + WD[j].off;
  };

  // one-time (per call) weight convert+transpose, amortized over all GEMMs
  for (int s = 0; s < 2; ++s)
    for (int i = 0; i < XDEPTH; ++i)
      for (int j = 0; j < 7; ++j) {
        const WDesc& d = WD[j];
        const float* src = prm[s][d.pidx] + (size_t)i * d.K * d.N;
        dim3 g(d.K / 32, d.N / 32);
        convert_transpose_kernel<<<g, 256, 0, stream>>>(src, wt(s, i, j), d.K,
                                                        d.N);
      }

  for (int i = 0; i < XDEPTH; ++i) {
    const size_t oV = (size_t)i * XDIM;

    auto self_attn = [&](int s, float* x) {
      launch_ln(x, prm[s][LN1G] + oV, prm[s][LN1B] + oV, lnq, stream);
      launch_gemm(lnq, XDIM, wt(s, i, 0), XDIM, nullptr, nullptr, nullptr, qkv,
                  3 * XDIM, 3 * XDIM, XDIM, 0, stream);
      launch_attn(qkv, 3 * XDIM, qkv + XDIM, 3 * XDIM, qkv + 2 * XDIM, 3 * XDIM,
                  attn, stream);
      launch_gemm(attn, XDIM, wt(s, i, 1), XDIM, prm[s][WOB] + oV, x, x,
                  nullptr, XDIM, XDIM, XDIM, 0, stream);
    };
    auto cross_attn = [&](int s, const float* xq, const float* xkv,
                          const float* resid, float* dst) {
      launch_ln(xq,  prm[s][LN2G] + oV, prm[s][LN2B] + oV, lnq,  stream);
      launch_ln(xkv, prm[s][LN2G] + oV, prm[s][LN2B] + oV, lnkv, stream);
      launch_gemm(lnq, XDIM, wt(s, i, 2), XDIM, nullptr, nullptr, nullptr, qkv,
                  XDIM, XDIM, XDIM, 0, stream);
      launch_gemm(lnkv, XDIM, wt(s, i, 3), XDIM, nullptr, nullptr, nullptr, kvb,
                  2 * XDIM, 2 * XDIM, XDIM, 0, stream);
      launch_attn(qkv, XDIM, kvb, 2 * XDIM, kvb + XDIM, 2 * XDIM, attn, stream);
      launch_gemm(attn, XDIM, wt(s, i, 4), XDIM, prm[s][W2OB] + oV, resid, dst,
                  nullptr, XDIM, XDIM, XDIM, 0, stream);
    };
    auto ff = [&](int s, const float* src, const float* resid, float* dst) {
      launch_ln(src, prm[s][LN3G] + oV, prm[s][LN3B] + oV, lnq, stream);
      launch_gemm(lnq, XDIM, wt(s, i, 5), XDIM, prm[s][BF1m] + (size_t)i * XMLP,
                  nullptr, nullptr, hid, XMLP, XMLP, XDIM, 1, stream);
      launch_gemm(hid, XMLP, wt(s, i, 6), XMLP, prm[s][BF2m] + oV, resid, dst,
                  nullptr, XDIM, XDIM, XMLP, 0, stream);
    };

    self_attn(0, xA);
    self_attn(1, xB);
    cross_attn(0, xA, xB, xA, xA2);   // xA2 = cross(lnA2(xA), lnA2(xB)) + xA
    cross_attn(1, xB, xA, xB, xB);    // uses post-self-attn xA (NOT xA2)
    ff(0, xA2, xA2, xA);              // xA = ff(ln(xA2)) + xA2
    ff(1, xB, xB, xB);                // xB = ff(ln(xB)) + xB
  }

  hipMemcpyAsync(d_out, xA, TD * 4, hipMemcpyDeviceToDevice, stream);
  hipMemcpyAsync((char*)d_out + TD * 4, xB, TD * 4, hipMemcpyDeviceToDevice,
                 stream);
}
